// SAGENodeRegressor_11888469475716
// MI455X (gfx1250) — matrix operand order repaired and checked by reading, not compiled, not run on hardware
//
#include <hip/hip_runtime.h>
#include <hip/hip_bf16.h>

typedef __attribute__((ext_vector_type(16))) __bf16 v16bf;
typedef __attribute__((ext_vector_type(4)))  __bf16 v4bf;
typedef __attribute__((ext_vector_type(8)))  float  v8f;

// ---------------- utility kernels ----------------

__global__ void zero_f32(float* __restrict__ p, int n) {
    int i = blockIdx.x * blockDim.x + threadIdx.x;
    if (i < n) p[i] = 0.0f;
}

// f32 -> bf16, 4 elements per thread (float4 load, 8B store)
__global__ void cvt_to_bf16(const float* __restrict__ in, __bf16* __restrict__ outp, int n4) {
    int i = blockIdx.x * blockDim.x + threadIdx.x;
    if (i >= n4) return;
    float4 v = ((const float4*)in)[i];
    v4bf o;
    o[0] = (__bf16)v.x; o[1] = (__bf16)v.y; o[2] = (__bf16)v.z; o[3] = (__bf16)v.w;
    *(v4bf*)(outp + (size_t)i * 4) = o;
}

__global__ void deg_accum(const int* __restrict__ dst, float* __restrict__ deg, int nE) {
    int e = blockIdx.x * blockDim.x + threadIdx.x;
    if (e < nE) atomicAdd(&deg[dst[e]], 1.0f);
}

__global__ void deg_inv(float* __restrict__ deg, int n) {
    int i = blockIdx.x * blockDim.x + threadIdx.x;
    if (i < n) deg[i] = 1.0f / fmaxf(deg[i], 1.0f);
}

// Pack a 128x128 f32 weight (row = K, col = N) into bf16, fragment-major:
// Wp[((ntile*4 + kk)*32 + lane)*16 + m] = W[kk*32 + (m>>3)*16 + (lane>>4)*8 + (m&7)][ntile*16 + (lane&15)]
// so each lane's entire 16-element B fragment is one contiguous 32-byte load.
__global__ void pack_weight_bf16(const float* __restrict__ W, __bf16* __restrict__ Wp) {
    int gid = blockIdx.x * blockDim.x + threadIdx.x;   // 0..16383
    if (gid >= 128 * 128) return;
    int m     = gid & 15;
    int lane  = (gid >> 4) & 31;
    int kk    = (gid >> 9) & 3;
    int ntile = (gid >> 11) & 7;
    int k = kk * 32 + (m >> 3) * 16 + (lane >> 4) * 8 + (m & 7);
    int n = ntile * 16 + (lane & 15);
    Wp[gid] = (__bf16)W[k * 128 + n];
}

// One 32-lane wave per edge: lane gathers 4 bf16 (8B load), accumulates into
// f32 agg with 4x GLOBAL_ATOMIC_ADD_F32. h/agg fit in 192MB L2 -> L2-resident.
__global__ void scatter_add(const __bf16* __restrict__ h,
                            const int* __restrict__ src,
                            const int* __restrict__ dst,
                            float* __restrict__ agg, int nE) {
    int gid  = blockIdx.x * blockDim.x + threadIdx.x;
    int e    = gid >> 5;
    int lane = gid & 31;
    if (e >= nE) return;
    int s = src[e];
    int d = dst[e];
    const v4bf v = *(const v4bf*)(h + (size_t)s * 128 + lane * 4);
    float* ap = agg + (size_t)d * 128 + lane * 4;
    atomicAdd(ap + 0, (float)v[0]);
    atomicAdd(ap + 1, (float)v[1]);
    atomicAdd(ap + 2, (float)v[2]);
    atomicAdd(ap + 3, (float)v[3]);
}

// ---------------- WMMA SAGE layer ----------------
// Block = 256 threads = 8 waves, 32 nodes x 128 out-dims per block.
// Wave w owns N-tile [16w,16w+16) and two M-tiles; each pre-packed bf16 B
// fragment (one 32B load) feeds two WMMAs. K=128 -> 4 K-steps per weight;
// agg@Wl and h@Wr accumulate into the same C tiles. h in/out is bf16.
__global__ void __launch_bounds__(256)
sage_layer(const float* __restrict__ agg, const float* __restrict__ invdeg,
           const __bf16* __restrict__ Hin, const __bf16* __restrict__ Wlp,
           const float* __restrict__ bl,  const __bf16* __restrict__ Wrp,
           __bf16* __restrict__ Hout, int n_nodes) {
    __shared__ __bf16 As[32 * 128];   // mean-aggregated neighbors, bf16
    __shared__ __bf16 Hs[32 * 128];   // root features, bf16

    const int node0 = blockIdx.x * 32;
    const int t = threadIdx.x;

    // Stage tiles: each thread owns 16 contiguous elements (one row segment).
    {
        int el0  = t * 16;                    // 0..4080, 16-aligned
        int r    = el0 >> 7;                  // 0..31
        int col0 = el0 & 127;
        int row  = node0 + r;
        if (row > n_nodes - 1) row = n_nodes - 1;    // clamp ragged last block
        // Hs: straight bf16 copy (32B global load -> LDS store)
        *(v16bf*)(Hs + el0) = *(const v16bf*)(Hin + (size_t)row * 128 + col0);
        // As: f32 segment-sum * inv_deg -> bf16
        float s = invdeg[row];
        const float* ap = agg + (size_t)row * 128 + col0;
        v16bf av;
        #pragma unroll
        for (int q = 0; q < 16; ++q) av[q] = (__bf16)(ap[q] * s);
        *(v16bf*)(As + el0) = av;
    }
    __syncthreads();

    const int wave = t >> 5;
    const int lane = t & 31;
    // A-fragment mapping (ISA 7.12.2, 16-bit A 16x32): per lane two 16B runs.
    const int arow = lane & 15;
    const int akb  = (lane >> 4) * 8;

    v8f c0 = {};   // rows node0 + 0..15
    v8f c1 = {};   // rows node0 + 16..31

    #pragma unroll
    for (int half = 0; half < 2; ++half) {
        const __bf16* S  = half ? Hs : As;
        const __bf16* Wp = half ? Wrp : Wlp;
        #pragma unroll
        for (int kk = 0; kk < 4; ++kk) {
            // B fragment: one contiguous 32-byte load per lane (pre-packed).
            v16bf b = *(const v16bf*)(Wp + ((size_t)(wave * 4 + kk) * 32 + lane) * 16);
            v16bf a0, a1;
            #pragma unroll
            for (int j = 0; j < 8; ++j) {
                int k0 = (j < 4) ? (akb + 2 * j) : (akb + 16 + 2 * (j - 4));
                int idx = kk * 32 + k0;
                a0[2 * j]     = S[arow * 128 + idx];
                a0[2 * j + 1] = S[arow * 128 + idx + 1];
                a1[2 * j]     = S[(arow + 16) * 128 + idx];
                a1[2 * j + 1] = S[(arow + 16) * 128 + idx + 1];
            }
            c0 = __builtin_amdgcn_wmma_f32_16x16x32_bf16(false, a0, false, b,
                                                         (short)0, c0, false, false);
            c1 = __builtin_amdgcn_wmma_f32_16x16x32_bf16(false, a1, false, b,
                                                         (short)0, c1, false, false);
        }
    }

    // Epilogue: bias + ReLU -> bf16. C/D layout: VGPR v -> row v + 8*(lane>>4).
    const int   ncol  = wave * 16 + (lane & 15);
    const float bias  = bl[ncol];
    const int   rbase = (lane >> 4) * 8;
    #pragma unroll
    for (int v = 0; v < 8; ++v) {
        int row0 = node0 + rbase + v;
        if (row0 < n_nodes)
            Hout[(size_t)row0 * 128 + ncol] = (__bf16)fmaxf(c0[v] + bias, 0.0f);
        int row1 = row0 + 16;
        if (row1 < n_nodes)
            Hout[(size_t)row1 * 128 + ncol] = (__bf16)fmaxf(c1[v] + bias, 0.0f);
    }
}

// ---------------- output GEMV ----------------
// One wave per node: lane holds 4 bf16 h-values + 4 f32 weights, shuffle-reduce.
__global__ void out_gemv(const __bf16* __restrict__ H, const float* __restrict__ Wout,
                         const float* __restrict__ bout, float* __restrict__ out, int n) {
    int node = blockIdx.x * (blockDim.x >> 5) + (threadIdx.x >> 5);
    int lane = threadIdx.x & 31;
    if (node >= n) return;
    const v4bf   hv = *(const v4bf*)(H + (size_t)node * 128 + lane * 4);
    const float4 wv = ((const float4*)Wout)[lane];
    float s = (float)hv[0] * wv.x + (float)hv[1] * wv.y +
              (float)hv[2] * wv.z + (float)hv[3] * wv.w;
    #pragma unroll
    for (int off = 16; off > 0; off >>= 1)
        s += __shfl_xor(s, off, 32);
    if (lane == 0) out[node] = s + bout[0];
}

// ---------------- launcher ----------------

extern "C" void kernel_launch(void* const* d_in, const int* in_sizes, int n_in,
                              void* d_out, int out_size, void* d_ws, size_t ws_size,
                              hipStream_t stream) {
    const float* x     = (const float*)d_in[0];
    const int*   ei    = (const int*)  d_in[1];
    const float* W_l   = (const float*)d_in[2];
    const float* b_l   = (const float*)d_in[3];
    const float* W_r   = (const float*)d_in[4];
    const float* W_out = (const float*)d_in[5];
    const float* b_out = (const float*)d_in[6];
    float*       out   = (float*)d_out;

    const int D = 128;
    const int N = in_sizes[0] / D;        // 50000
    const int E = in_sizes[1] / 2;        // 600000
    const int L = in_sizes[3] / D;        // 3
    const int* src = ei;
    const int* dst = ei + E;

    // workspace layout
    __bf16* hx  = (__bf16*)d_ws;          // bf16 copy of x
    __bf16* h0  = hx + (size_t)N * D;
    __bf16* h1  = h0 + (size_t)N * D;
    float*  agg = (float*)(h1 + (size_t)N * D);
    float*  deg = agg + (size_t)N * D;
    __bf16* wpk = (__bf16*)(deg + N);     // 6 packed 128x128 bf16 matrices

    const int TB  = 256;
    const int WSZ = D * D;                // 16384 elements per matrix

    // Pre-pack all layer weights into bf16 fragment-major layout.
    for (int l = 0; l < L; ++l) {
        pack_weight_bf16<<<(WSZ + TB - 1) / TB, TB, 0, stream>>>(
            W_l + (size_t)l * WSZ, wpk + (size_t)(2 * l + 0) * WSZ);
        pack_weight_bf16<<<(WSZ + TB - 1) / TB, TB, 0, stream>>>(
            W_r + (size_t)l * WSZ, wpk + (size_t)(2 * l + 1) * WSZ);
    }

    // x -> bf16 feature table
    cvt_to_bf16<<<(N * D / 4 + TB - 1) / TB, TB, 0, stream>>>(x, hx, N * D / 4);

    // inverse mean-degree
    zero_f32 <<<(N + TB - 1) / TB, TB, 0, stream>>>(deg, N);
    deg_accum<<<(E + TB - 1) / TB, TB, 0, stream>>>(dst, deg, E);
    deg_inv  <<<(N + TB - 1) / TB, TB, 0, stream>>>(deg, N);

    const __bf16* hin = hx;
    __bf16* bufs[2] = {h0, h1};
    for (int l = 0; l < L; ++l) {
        __bf16* hout = bufs[l & 1];
        zero_f32   <<<(N * D + TB - 1) / TB, TB, 0, stream>>>(agg, N * D);
        scatter_add<<<(E * 32 + TB - 1) / TB, TB, 0, stream>>>(hin, src, dst, agg, E);
        sage_layer <<<(N + 31) / 32, TB, 0, stream>>>(
            agg, deg, hin,
            wpk + (size_t)(2 * l + 0) * WSZ, b_l + (size_t)l * D,
            wpk + (size_t)(2 * l + 1) * WSZ,
            hout, N);
        hin = hout;
    }

    out_gemv<<<(N + 7) / 8, TB, 0, stream>>>(hin, W_out, b_out, out, N);
}
